// GINModel3Layers_27290222199187
// MI455X (gfx1250) — matrix operand order repaired and checked by reading, not compiled, not run on hardware
//
#include <hip/hip_runtime.h>

// ---------------------------------------------------------------------------
// GIN (3 layers) triplet forward for MI455X (gfx1250, wave32, WMMA).
// GEMMs: v_wmma_f32_16x16x32_bf16 (bf16 in, f32 accumulate).
// Edge aggregation: f32 global atomics (L2-resident: 25.6MB << 192MB L2).
// ---------------------------------------------------------------------------

typedef __attribute__((ext_vector_type(16))) __bf16 v16bf;
typedef __attribute__((ext_vector_type(8)))  __bf16 v8bf;
typedef __attribute__((ext_vector_type(8)))  float  v8f;

#define NODES 50000
#define EDGES 800000
#define NG    128
#define HID   128

// ---------------- elementwise / aggregation kernels ----------------

__global__ void k_zero(float* __restrict__ p, int n) {
  int i = blockIdx.x * blockDim.x + threadIdx.x;
  if (i < n) p[i] = 0.f;
}

// layer-1 aggregation: x is [N,1] scalar per node
__global__ void k_edge_scalar(const float* __restrict__ x,
                              const long long* __restrict__ ei,
                              float* __restrict__ aggr) {
  int e = blockIdx.x * blockDim.x + threadIdx.x;
  if (e >= EDGES) return;
  int src = (int)ei[e];
  int dst = (int)ei[EDGES + e];
  atomicAdd(&aggr[dst], x[src]);
}

// layer-1 first MLP stage: t[i][j] = relu((x[i]+aggr[i])*w1[j] + b1[j]) -> bf16
__global__ void k_layer1(const float* __restrict__ x,
                         const float* __restrict__ aggr,
                         const float* __restrict__ w1,
                         const float* __restrict__ b1,
                         __bf16* __restrict__ A) {
  int t = blockIdx.x * blockDim.x + threadIdx.x;
  if (t >= NODES * HID) return;
  int i = t >> 7, j = t & 127;
  float h = x[i] + aggr[i];
  float v = fmaf(h, w1[j], b1[j]);
  A[t] = (__bf16)(v > 0.f ? v : 0.f);
}

// weights: f32 [K,N] row-major -> bf16 transposed [N,K] (contiguous B frags)
__global__ void k_transpose_w(const float* __restrict__ w, __bf16* __restrict__ wT) {
  int t = blockIdx.x * blockDim.x + threadIdx.x;  // 16384
  int n = t >> 7, k = t & 127;
  wT[t] = (__bf16)w[k * HID + n];
}

// vector segment-sum: one wave per edge, 4 floats per lane
__global__ void k_edge_vec(const float* __restrict__ X,
                           const long long* __restrict__ ei,
                           float* __restrict__ aggr) {
  long long t = (long long)blockIdx.x * blockDim.x + threadIdx.x;
  int e = (int)(t >> 5);
  if (e >= EDGES) return;
  int j = ((int)t & 31) << 2;
  int src = (int)ei[e];
  int dst = (int)ei[EDGES + e];
  const float4 v = *(const float4*)(X + (size_t)src * HID + j);
  float* p = aggr + (size_t)dst * HID + j;
  atomicAdd(p + 0, v.x);
  atomicAdd(p + 1, v.y);
  atomicAdd(p + 2, v.z);
  atomicAdd(p + 3, v.w);
}

// h = x + aggr, convert to bf16 GEMM operand
__global__ void k_finalize(const float* __restrict__ X, const float* __restrict__ AG,
                           __bf16* __restrict__ A) {
  int t = blockIdx.x * blockDim.x + threadIdx.x;
  if (t >= NODES * HID) return;
  A[t] = (__bf16)(X[t] + AG[t]);
}

__global__ void k_pool(const float* __restrict__ X, const long long* __restrict__ batch,
                       float* __restrict__ pooled) {
  int t = blockIdx.x * blockDim.x + threadIdx.x;
  if (t >= NODES * HID) return;
  int i = t >> 7, j = t & 127;
  atomicAdd(&pooled[(int)batch[i] * HID + j], X[t]);
}

__global__ void k_tobf16(const float* __restrict__ s, __bf16* __restrict__ d, int n) {
  int t = blockIdx.x * blockDim.x + threadIdx.x;
  if (t < n) d[t] = (__bf16)s[t];
}

// ---------------- WMMA GEMM: [M,128] x [128,128] + bias (+relu) ----------------
// One wave = one 16-row output tile; sweeps 8 N-tiles; K=128 = 4 wmma steps.
// A frag (16x32 bf16): lanes 0-15 -> M=lane, K 0..15; lanes 16-31 -> K 16..31
// B frag (32x16 bf16): lane n=lane&15 holds column n; lanes 0-15 K0-15, lanes 16-31 K16-31
// D (16x16 f32): VGPR r -> M = (lane>=16 ? 8 : 0)+r, N = lane&15

union V16 { v16bf v; v8bf h[2]; };

template <int OUT_BF16, int RELU>
__global__ void k_gemm128(const __bf16* __restrict__ A, const __bf16* __restrict__ WT,
                          const float* __restrict__ bias, void* __restrict__ out, int M) {
  const int wave = threadIdx.x >> 5;
  const int lane = threadIdx.x & 31;
  const int tile = blockIdx.x * (blockDim.x >> 5) + wave;
  if (tile * 16 >= M) return;  // wave-uniform: EXEC stays all-ones
  const int m  = lane & 15;
  const int hi = lane >> 4;

  const __bf16* arow = A + (size_t)(tile * 16 + m) * HID;
  V16 a[4];
#pragma unroll
  for (int kb = 0; kb < 4; ++kb) {
    a[kb].h[0] = *(const v8bf*)(arow + kb * 32 + hi * 16);
    a[kb].h[1] = *(const v8bf*)(arow + kb * 32 + hi * 16 + 8);
  }

#pragma unroll
  for (int nt = 0; nt < 8; ++nt) {
    const __bf16* brow = WT + (size_t)(nt * 16 + m) * HID;  // column nt*16+m of W
    v8f c = {};
#pragma unroll
    for (int kb = 0; kb < 4; ++kb) {
      v16bf b = *(const v16bf*)(brow + kb * 32 + hi * 16);
      c = __builtin_amdgcn_wmma_f32_16x16x32_bf16(false, a[kb].v, false, b,
                                                  (short)0, c, false, false);
    }
    const float bn = bias[nt * 16 + m];
#pragma unroll
    for (int r = 0; r < 8; ++r) {
      float v = c[r] + bn;
      if (RELU) v = v > 0.f ? v : 0.f;
      size_t idx = (size_t)(tile * 16 + hi * 8 + r) * HID + (size_t)(nt * 16 + m);
      if (OUT_BF16) ((__bf16*)out)[idx] = (__bf16)v;
      else          ((float*)out)[idx] = v;
    }
  }
}

// ---------------- host launch ----------------

extern "C" void kernel_launch(void* const* d_in, const int* in_sizes, int n_in,
                              void* d_out, int out_size, void* d_ws, size_t ws_size,
                              hipStream_t stream) {
  (void)in_sizes; (void)n_in; (void)out_size; (void)ws_size;
  char* ws = (char*)d_ws;
  size_t off = 0;
  auto alloc = [&](size_t bytes) -> void* {
    void* p = ws + off;
    off = (off + bytes + 255) & ~(size_t)255;
    return p;
  };

  // workspace layout (~64.5 MB)
  __bf16* wT[6];
  const int widx[6] = {11, 13, 15, 17, 19, 21};  // c1w2, c2w1, c2w2, c3w1, c3w2, fc_w
  for (int i = 0; i < 6; ++i) wT[i] = (__bf16*)alloc((size_t)HID * HID * sizeof(__bf16));
  float*  aggrS   = (float*)alloc((size_t)NODES * sizeof(float));
  __bf16* A       = (__bf16*)alloc((size_t)NODES * HID * sizeof(__bf16));
  float*  X       = (float*)alloc((size_t)NODES * HID * sizeof(float));
  float*  AG      = (float*)alloc((size_t)NODES * HID * sizeof(float));
  __bf16* T       = (__bf16*)AG;  // alias: AG dead once finalize has consumed it
  float*  pooled  = (float*)alloc((size_t)NG * HID * sizeof(float));
  __bf16* pooledb = (__bf16*)alloc((size_t)NG * HID * sizeof(__bf16));

  // one-time weight prep: f32 -> bf16 transposed
  for (int i = 0; i < 6; ++i)
    k_transpose_w<<<(HID * HID) / 256, 256, 0, stream>>>((const float*)d_in[widx[i]], wT[i]);

  const int NH = NODES * HID;
  const int nhBlocks = (NH + 255) / 256;
  const int tiles = NODES / 16;            // 3125 (50000 % 16 == 0)
  const int gemmBlocks = (tiles + 7) / 8;  // 8 waves per block
  const int edgeVecBlocks = (EDGES * 32 + 255) / 256;

  const float* c1w1 = (const float*)d_in[9];
  const float* c1b1 = (const float*)d_in[10];
  const float* c1b2 = (const float*)d_in[12];
  const float* c2b1 = (const float*)d_in[14];
  const float* c2b2 = (const float*)d_in[16];
  const float* c3b1 = (const float*)d_in[18];
  const float* c3b2 = (const float*)d_in[20];
  const float* fcb  = (const float*)d_in[22];

  for (int g = 0; g < 3; ++g) {
    const float*     x     = (const float*)d_in[3 * g + 0];
    const long long* ei    = (const long long*)d_in[3 * g + 1];
    const long long* batch = (const long long*)d_in[3 * g + 2];
    float* outg = (float*)d_out + (size_t)g * NG * HID;

    // ---- layer 1 (scalar features) ----
    k_zero<<<(NODES + 255) / 256, 256, 0, stream>>>(aggrS, NODES);
    k_edge_scalar<<<(EDGES + 255) / 256, 256, 0, stream>>>(x, ei, aggrS);
    k_layer1<<<nhBlocks, 256, 0, stream>>>(x, aggrS, c1w1, c1b1, A);
    k_gemm128<0, 1><<<gemmBlocks, 256, 0, stream>>>(A, wT[0], c1b2, X, NODES);  // X = x1

    // ---- layers 2, 3 ----
    for (int L = 0; L < 2; ++L) {
      k_zero<<<nhBlocks, 256, 0, stream>>>(AG, NH);
      k_edge_vec<<<edgeVecBlocks, 256, 0, stream>>>(X, ei, AG);
      k_finalize<<<nhBlocks, 256, 0, stream>>>(X, AG, A);  // A = bf16(X + AG)
      k_gemm128<1, 1><<<gemmBlocks, 256, 0, stream>>>(
          A, wT[1 + 2 * L], (L == 0) ? c2b1 : c3b1, T, NODES);        // T = relu(A@w1+b1) bf16
      k_gemm128<0, 1><<<gemmBlocks, 256, 0, stream>>>(
          T, wT[2 + 2 * L], (L == 0) ? c2b2 : c3b2, X, NODES);        // X = relu(T@w2+b2)
    }

    // ---- pooling + fc ----
    k_zero<<<(NG * HID + 255) / 256, 256, 0, stream>>>(pooled, NG * HID);
    k_pool<<<nhBlocks, 256, 0, stream>>>(X, batch, pooled);
    k_tobf16<<<(NG * HID + 255) / 256, 256, 0, stream>>>(pooled, pooledb, NG * HID);
    k_gemm128<0, 0><<<1, 256, 0, stream>>>(pooledb, wT[5], fcb, outg, NG);
  }
}
